// SDERNNCell_58600533787376
// MI455X (gfx1250) — matrix operand order repaired and checked by reading, not compiled, not run on hardware
//
#include <hip/hip_runtime.h>
#include <math.h>

// ---------------------------------------------------------------------------
// SDE Euler-Maruyama with 4 tiny MLPs, mapped onto CDNA5 WMMA (gfx1250).
//
// Layout strategy (all GEMMs transposed so trajectory = N/column dim):
//   layer1: hT = A(W1^T, M=hidden,K=feat64) x B(featsT, K=feat64,N=traj16)
//   layer2: oT = A(W2^T, M=out16, K=hid64)  x B(hT,     K=hid64, N=traj16)
// C/D layout: lane owns trajectory (lane%16); VGPR r holds row M = r + 8*(lane>=16).
// Feature ordering (K): [0..15]=Re(x), [16..31]=Im(x), [32]=t, [33..37]=params,
// [38]=1.0 (folds B1 into the GEMM), [39..63]=0 pad.
// tanh uses the CDNA5 hardware V_TANH_F32 transcendental.
// Per-step LDS transposes are wave-private -> wave-level s_wait_dscnt fence
// instead of block barriers.
// ---------------------------------------------------------------------------

namespace {
constexpr int   kB       = 128;
constexpr int   kTrajRep = 128;
constexpr int   kSteps   = 63;
constexpr int   kN       = kB * kTrajRep;       // 16384 trajectories
constexpr float kDt      = 0.00390625f;         // 2^-8
constexpr int   kWaves   = 2;                   // waves per block
constexpr int   kThreads = kWaves * 32;
constexpr int   kW1Halves = 32 * 512;           // 32 fragments * 512 halves
constexpr int   kW1Bytes  = kW1Halves * 2;      // 32 KiB swizzled W1(+B1)
constexpr int   kStageBytes = 2048 + 4 * 2048;  // feats(16x64 f16) + h[4](16x64 f16)
constexpr int   kSmemBytes  = kW1Bytes + kWaves * kStageBytes;  // 52 KiB
}

typedef _Float16 v16h __attribute__((ext_vector_type(16)));
typedef _Float16 hv8  __attribute__((ext_vector_type(8)));
typedef float    v8f  __attribute__((ext_vector_type(8)));
typedef float    f4   __attribute__((ext_vector_type(4)));

// Hardware tanh (TRANS32 class on CDNA5; co-executes with WMMA).
#if __has_builtin(__builtin_amdgcn_tanhf)
static __device__ __forceinline__ float hw_tanh(float x) {
  return __builtin_amdgcn_tanhf(x);
}
#else
static __device__ __forceinline__ float hw_tanh(float x) {
  float y;
  // v_nop covers the TRANS32 result-use delay slot (ISA §7.4 case 3).
  asm volatile("v_tanh_f32 %0, %1\n\tv_nop" : "=v"(y) : "v"(x));
  return y;
}
#endif

// Wave-local LDS fence: DS ops of a wave complete in order, so after
// s_wait_dscnt 0 all lanes' prior ds_stores are visible to all lanes' loads.
static __device__ __forceinline__ void wave_lds_fence() {
  __builtin_amdgcn_wave_barrier();
  asm volatile("s_wait_dscnt 0" ::: "memory");
  __builtin_amdgcn_wave_barrier();
}

extern "C" __global__ __launch_bounds__(kThreads, 1)
void sde_em_wmma(const float* __restrict__ x_real,
                 const float* __restrict__ x_imag,
                 const float* __restrict__ params,
                 const float* __restrict__ noise,
                 const float* __restrict__ W1,
                 const float* __restrict__ B1,
                 const float* __restrict__ W2,
                 const float* __restrict__ B2,
                 float* __restrict__ out) {
  extern __shared__ __align__(32) char smem[];
  _Float16* w1s = (_Float16*)smem;

  const int tid  = threadIdx.x;
  const int lane = tid & 31;
  const int wave = tid >> 5;
  const int hh   = lane >> 4;    // 0: lanes 0-15, 1: lanes 16-31
  const int ln16 = lane & 15;

  _Float16* feats = (_Float16*)(smem + kW1Bytes + wave * kStageBytes);
  _Float16* hbuf  = feats + 1024;

  // ---- one-time per block: swizzle W1 (with B1 folded at K=38) into LDS,
  // fragment-linear so each lane later reads its 32B A-fragment contiguously.
  // frag = (net*2 + kc)*4 + mt ; per-lane halves j: K = kc*32 + 16*(lane>=16)
  //                                                  + j
  for (int idx = tid; idx < kW1Halves; idx += kThreads) {
    int frag = idx >> 9;
    int rem  = idx & 511;
    int lln  = rem >> 4;
    int j    = rem & 15;
    int net  = frag >> 3;
    int kc   = (frag >> 2) & 1;
    int mt   = frag & 3;
    int K    = kc * 32 + ((lln >> 4) << 4) + j;
    int M    = mt * 16 + (lln & 15);        // hidden unit index
    float v  = 0.0f;
    if (K < 32)       v = W1[net * (38 * 64) + (K + 1) * 64 + M];  // Re/Im feats
    else if (K == 32) v = W1[net * (38 * 64) + M];                 // t feature
    else if (K < 38)  v = W1[net * (38 * 64) + K * 64 + M];        // params
    else if (K == 38) v = B1[net * 64 + M];                        // bias via 1.0
    w1s[idx] = (_Float16)v;
  }

  // ---- W2^T A-fragments held in registers for all 63 steps (8 x v16h)
  v16h w2f[8];
#pragma unroll
  for (int f = 0; f < 8; ++f) {
    const int net = f >> 1, kc = f & 1;
#pragma unroll
    for (int j = 0; j < 16; ++j) {
      const int K = kc * 32 + hh * 16 + j;
      w2f[f][j] = (_Float16)W2[(net * 64 + K) * 16 + ln16];
    }
  }

  // ---- B2 bias preloaded in C-layout (row M = r + 8*hh)
  v8f b2a[4];
#pragma unroll
  for (int net = 0; net < 4; ++net)
#pragma unroll
    for (int r = 0; r < 8; ++r)
      b2a[net][r] = B2[net * 16 + hh * 8 + r];

  // ---- per-wave trajectory tile; lane owns trajectory = tile*16 + lane%16
  const int tile = blockIdx.x * kWaves + wave;
  const int traj = tile * 16 + ln16;
  const int bidx = traj & (kB - 1);   // tiled initial state: traj % B

  v8f xre, xim;
#pragma unroll
  for (int r = 0; r < 8; ++r) {
    xre[r] = x_real[bidx * 16 + hh * 8 + r];
    xim[r] = x_imag[bidx * 16 + hh * 8 + r];
  }

  const _Float16 q0 = (_Float16)params[0];
  const _Float16 q1 = (_Float16)params[1];
  const _Float16 q2 = (_Float16)params[2];
  const _Float16 q3 = (_Float16)params[3];
  const _Float16 q4 = (_Float16)params[4];

  // ---- zero the constant K=40..63 feature padding once
  if (lane < 16) {
    hv8 z;
#pragma unroll
    for (int r = 0; r < 8; ++r) z[r] = (_Float16)0.0f;
    _Float16* row = feats + lane * 64;
    *(hv8*)(row + 40) = z;
    *(hv8*)(row + 48) = z;
    *(hv8*)(row + 56) = z;
  }
  __syncthreads();   // w1s is shared by both waves of the block

  // =================== 63 Euler-Maruyama steps ===================
#pragma unroll 1
  for (int step = 0; step < kSteps; ++step) {
    const float t = (float)step * kDt;

    // stage featsT row for this lane's trajectory (packed f16, b128 stores)
    hv8 re, im;
#pragma unroll
    for (int r = 0; r < 8; ++r) { re[r] = (_Float16)xre[r]; im[r] = (_Float16)xim[r]; }
    _Float16* frow = feats + ln16 * 64;
    *(hv8*)(frow + hh * 8)       = re;   // K = 0..15  (Re)
    *(hv8*)(frow + 16 + hh * 8)  = im;   // K = 16..31 (Im)
    if (lane < 16) {
      hv8 tp;
      tp[0] = (_Float16)t; tp[1] = q0; tp[2] = q1; tp[3] = q2;
      tp[4] = q3; tp[5] = q4; tp[6] = (_Float16)1.0f; tp[7] = (_Float16)0.0f;
      *(hv8*)(frow + 32) = tp;           // K = 32..39 (t, params, bias-one)
    }
    wave_lds_fence();                    // staging is wave-private

    // B fragments of featsT: lane column = trajectory, contiguous 32B per chunk
    const v16h bf0 = *(const v16h*)(feats + ln16 * 64 + hh * 16);
    const v16h bf1 = *(const v16h*)(feats + ln16 * 64 + 32 + hh * 16);

    // ---- layer 1: 32 WMMAs -> v_tanh_f32 -> pack f16 -> hbuf (row-major/traj)
#pragma unroll
    for (int net = 0; net < 4; ++net) {
#pragma unroll
      for (int mt = 0; mt < 4; ++mt) {
        const int frag0 = (net * 2) * 4 + mt;
        const v16h a0 = *(const v16h*)(w1s + frag0 * 512 + lane * 16);
        const v16h a1 = *(const v16h*)(w1s + (frag0 + 4) * 512 + lane * 16);
        v8f acc = {};
        acc = __builtin_amdgcn_wmma_f32_16x16x32_f16(false, a0, false, bf0,
                                                     (short)0, acc, false, false);
        acc = __builtin_amdgcn_wmma_f32_16x16x32_f16(false, a1, false, bf1,
                                                     (short)0, acc, false, false);
        hv8 hp;
#pragma unroll
        for (int r = 0; r < 8; ++r) hp[r] = (_Float16)hw_tanh(acc[r]);
        *(hv8*)(hbuf + net * 1024 + ln16 * 64 + mt * 16 + hh * 8) = hp;
      }
    }
    wave_lds_fence();

    const float dw = noise[traj * kSteps + step];

    // ---- layer 2: 8 WMMAs (W2^T frags in VGPRs), then Euler-Maruyama update
#pragma unroll
    for (int net = 0; net < 4; ++net) {
      const v16h b0 = *(const v16h*)(hbuf + net * 1024 + ln16 * 64 + hh * 16);
      const v16h b1 = *(const v16h*)(hbuf + net * 1024 + ln16 * 64 + 32 + hh * 16);
      v8f acc = b2a[net];
      acc = __builtin_amdgcn_wmma_f32_16x16x32_f16(false, w2f[net * 2 + 0], false,
                                                   b0, (short)0, acc, false, false);
      acc = __builtin_amdgcn_wmma_f32_16x16x32_f16(false, w2f[net * 2 + 1], false,
                                                   b1, (short)0, acc, false, false);
      const float c = (net < 2) ? kDt : dw;   // nets 0,1: drift*dt; 2,3: diff*dW
      if ((net & 1) == 0) {
#pragma unroll
        for (int r = 0; r < 8; ++r) xre[r] += acc[r] * c;
      } else {
#pragma unroll
        for (int r = 0; r < 8; ++r) xim[r] += acc[r] * c;
      }
    }
  }

  // ---- write y = [Re(xT), Im(xT)] : out[traj][0..15]=re, [16..31]=im
  float* orow = out + traj * 32 + hh * 8;
  f4 r0, r1, i0, i1;
#pragma unroll
  for (int r = 0; r < 4; ++r) {
    r0[r] = xre[r]; r1[r] = xre[4 + r];
    i0[r] = xim[r]; i1[r] = xim[4 + r];
  }
  *(f4*)(orow)      = r0;
  *(f4*)(orow + 4)  = r1;
  *(f4*)(orow + 16) = i0;
  *(f4*)(orow + 20) = i1;
}

extern "C" void kernel_launch(void* const* d_in, const int* in_sizes, int n_in,
                              void* d_out, int out_size, void* d_ws, size_t ws_size,
                              hipStream_t stream) {
  const float* x_real = (const float*)d_in[0];
  const float* x_imag = (const float*)d_in[1];
  const float* params = (const float*)d_in[2];
  const float* noise  = (const float*)d_in[3];
  const float* W1     = (const float*)d_in[4];
  const float* B1     = (const float*)d_in[5];
  const float* W2     = (const float*)d_in[6];
  const float* B2     = (const float*)d_in[7];
  float* out = (float*)d_out;

  const int tiles = kN / 16;                 // 1024 wave tiles
  dim3 grid(tiles / kWaves);                 // 512 blocks
  dim3 block(kThreads);                      // 64 threads = 2 waves
  sde_em_wmma<<<grid, block, kSmemBytes, stream>>>(
      x_real, x_imag, params, noise, W1, B1, W2, B2, out);
}
